// RotaryTransformerILMModelWithClassification_90555090469288
// MI455X (gfx1250) — compile-verified, matrix operand order checked by reading, not run on hardware
//
#include <hip/hip_runtime.h>
#include <hip/hip_bf16.h>

#ifndef __has_builtin
#define __has_builtin(x) 0
#endif

// ---------------- CDNA5 specifics ----------------
typedef __attribute__((ext_vector_type(16))) __bf16 v16bf;
typedef __attribute__((ext_vector_type(8)))  __bf16 v8bf;
typedef __attribute__((ext_vector_type(8)))  float  v8f;
typedef int v4i_t __attribute__((ext_vector_type(4)));
typedef __attribute__((address_space(1))) v4i_t* as1_v4i_p;  // global
typedef __attribute__((address_space(3))) v4i_t* as3_v4i_p;  // LDS

#if __has_builtin(__builtin_amdgcn_global_load_async_to_lds_b128)
#define HAVE_ASYNC_LDS 1
#else
#define HAVE_ASYNC_LDS 0
#endif

// async 16-byte global -> LDS copy (8 bf16)
__device__ __forceinline__ void copy16_g2lds(__bf16* lds_dst, const __bf16* gsrc) {
#if HAVE_ASYNC_LDS
  __builtin_amdgcn_global_load_async_to_lds_b128(
      (as1_v4i_p)gsrc, (as3_v4i_p)lds_dst, 0, 0);
#else
  *(v8bf*)lds_dst = *(const v8bf*)gsrc;
#endif
}

__device__ __forceinline__ void wait_async_copies() {
#if HAVE_ASYNC_LDS
#if __has_builtin(__builtin_amdgcn_s_wait_asynccnt)
  __builtin_amdgcn_s_wait_asynccnt(0);
#else
  asm volatile("s_wait_asynccnt 0" ::: "memory");
#endif
#endif
}

// One 16x32 bf16 WMMA operand fragment from a bf16 LDS tile row.
// ISA 16-bit A/B layout: lanes 0-15 -> K {0..7,16..23}; lanes 16-31 -> K {8..15,24..31}.
__device__ __forceinline__ v16bf frag16(const __bf16* rowp, int khalf) {
  v8bf lo = *(const v8bf*)(rowp + khalf);
  v8bf hi = *(const v8bf*)(rowp + 16 + khalf);
  return __builtin_shufflevector(lo, hi, 0, 1, 2, 3, 4, 5, 6, 7,
                                 8, 9, 10, 11, 12, 13, 14, 15);
}

// ---------------- GEMM: Out[M,N] = A[M,K] * op(B) + bias ----------------
// TRANSB == false: B stored [N,K] (Out = A * B^T)  -> x @ W.T, Q @ K^T
// TRANSB == true : B stored [K,N] (Out = A * B)    -> attn @ V
// All matrix inputs bf16; accumulate fp32; store fp32 (OutF) and/or bf16 (OutH).
template <bool TRANSB, bool RELU>
__global__ __launch_bounds__(256)
void gemm_wmma(const __bf16* __restrict__ A, const __bf16* __restrict__ Bm,
               const float* __restrict__ bias, float* __restrict__ OutF,
               __bf16* __restrict__ OutH, int M, int N, int K,
               long long sA, long long sB, long long sO, float scale) {
  constexpr int BK = 32, LDT = 40;  // bf16 LDS stride (row = 80B, 16B aligned)
  __shared__ __bf16 smA[2][128 * LDT];
  __shared__ __bf16 smB[2][128 * LDT];

  const __bf16* Ab = A + (long long)blockIdx.z * sA;
  const __bf16* Bb = Bm + (long long)blockIdx.z * sB;

  const int m0 = blockIdx.y * 128, n0 = blockIdx.x * 128;
  const int tid = threadIdx.x, lane = tid & 31, wv = tid >> 5;
  const int wm = wv & 3, wn = wv >> 2;  // 4x2 waves -> 128x128 tile

  v8f acc[2][4] = {};

  // stage one 128x32 bf16 tile pair into LDS buffer bi (async when interior)
  auto stageAB = [&](int k0, int bi) {
    __bf16* tA = &smA[bi][0];
    __bf16* tB = &smB[bi][0];
    const bool intA = (m0 + 128 <= M) && (k0 + BK <= K);
#pragma unroll
    for (int it = 0; it < 2; ++it) {
      int idx = tid + it * 256;      // 512 x 16B transfers
      int r = idx >> 2, c = (idx & 3) * 8;
      __bf16* dst = &tA[r * LDT + c];
      if (intA) {
        copy16_g2lds(dst, &Ab[(long long)(m0 + r) * K + k0 + c]);
      } else {
#pragma unroll
        for (int j = 0; j < 8; ++j)
          dst[j] = (m0 + r < M && k0 + c + j < K)
                       ? Ab[(long long)(m0 + r) * K + k0 + c + j] : (__bf16)0.f;
      }
    }
    if (!TRANSB) {
      const bool intB = (n0 + 128 <= N) && (k0 + BK <= K);
#pragma unroll
      for (int it = 0; it < 2; ++it) {
        int idx = tid + it * 256;
        int r = idx >> 2, c = (idx & 3) * 8;
        __bf16* dst = &tB[r * LDT + c];
        if (intB) {
          copy16_g2lds(dst, &Bb[(long long)(n0 + r) * K + k0 + c]);
        } else {
#pragma unroll
          for (int j = 0; j < 8; ++j)
            dst[j] = (n0 + r < N && k0 + c + j < K)
                         ? Bb[(long long)(n0 + r) * K + k0 + c + j] : (__bf16)0.f;
        }
      }
    } else {
#pragma unroll
      for (int it = 0; it < 16; ++it) {
        int idx = tid + it * 256;    // 4096 scalar elements
        int n = idx >> 5, k = idx & 31;
        tB[n * LDT + k] = (n0 + n < N && k0 + k < K)
                              ? Bb[(long long)(k0 + k) * N + n0 + n] : (__bf16)0.f;
      }
    }
  };

  const int nsteps = (K + BK - 1) / BK;
  stageAB(0, 0);
  wait_async_copies();
  __syncthreads();

  const int rsel = lane & 15, khalf = (lane >> 4) * 8;
  for (int ks = 0; ks < nsteps; ++ks) {
    const int bi = ks & 1;
    if (ks + 1 < nsteps) stageAB((ks + 1) * BK, bi ^ 1);  // prefetch next tiles

    v16bf fa[2], fb[4];
#pragma unroll
    for (int rt = 0; rt < 2; ++rt)
      fa[rt] = frag16(&smA[bi][(wm * 32 + rt * 16 + rsel) * LDT], khalf);
#pragma unroll
    for (int ct = 0; ct < 4; ++ct)
      fb[ct] = frag16(&smB[bi][(wn * 64 + ct * 16 + rsel) * LDT], khalf);
#pragma unroll
    for (int rt = 0; rt < 2; ++rt)
#pragma unroll
      for (int ct = 0; ct < 4; ++ct)
        acc[rt][ct] = __builtin_amdgcn_wmma_f32_16x16x32_bf16(
            false, fa[rt], false, fb[ct], (short)0, acc[rt][ct], false, false);

    wait_async_copies();
    __syncthreads();
  }

  // store: C/D layout: VGPR i -> M=i (lanes 0-15) / M=i+8 (lanes 16-31), N=lane&15
  const int rbase = (lane >> 4) * 8, csel = lane & 15;
#pragma unroll
  for (int rt = 0; rt < 2; ++rt)
#pragma unroll
    for (int ct = 0; ct < 4; ++ct)
#pragma unroll
      for (int i = 0; i < 8; ++i) {
        int row = m0 + wm * 32 + rt * 16 + rbase + i;
        int col = n0 + wn * 64 + ct * 16 + csel;
        if (row < M && col < N) {
          float v = acc[rt][ct][i] * scale;
          if (bias) v += bias[col];
          if (RELU) v = fmaxf(v, 0.f);
          long long off = (long long)blockIdx.z * sO + (long long)row * N + col;
          if (OutF) OutF[off] = v;
          if (OutH) OutH[off] = (__bf16)v;
        }
      }
}

// ---------------- elementwise kernels ----------------
__global__ void cvt_bf16_kernel(const float* __restrict__ src, __bf16* __restrict__ dst,
                                long long n) {
  long long i = ((long long)blockIdx.x * 256 + threadIdx.x) * 4;
  if (i + 3 < n) {
    float4 v = *(const float4*)(src + i);
    dst[i] = (__bf16)v.x; dst[i + 1] = (__bf16)v.y;
    dst[i + 2] = (__bf16)v.z; dst[i + 3] = (__bf16)v.w;
  } else {
    for (int j = 0; j < 4; ++j)
      if (i + j < n) dst[i + j] = (__bf16)src[i + j];
  }
}

__global__ void embed_kernel(const int* __restrict__ xt, const float* __restrict__ emb,
                             float* __restrict__ X, __bf16* __restrict__ Xb,
                             long long total, int D) {
  long long i = (long long)blockIdx.x * 256 + threadIdx.x;
  if (i >= total) return;
  int t = (int)(i / D), d = (int)(i % D);
  float v = emb[(long long)xt[t] * D + d];
  X[i] = v;
  Xb[i] = (__bf16)v;
}

__global__ void rope_split_kernel(const __bf16* __restrict__ ql, const __bf16* __restrict__ kl,
                                  const __bf16* __restrict__ vl, __bf16* __restrict__ Q,
                                  __bf16* __restrict__ K, __bf16* __restrict__ V,
                                  int Bsz, int S, int H, int HD) {
  const int half = HD / 2;
  long long i = (long long)blockIdx.x * 256 + threadIdx.x;
  long long total = (long long)Bsz * S * H * half;
  if (i >= total) return;
  int r = (int)(i % half); long long t = i / half;
  int h = (int)(t % H); t /= H;
  int s = (int)(t % S); int b = (int)(t / S);
  const int Dm = H * HD;
  long long inoff = ((long long)(b * S + s)) * Dm + h * HD;
  long long outoff = (((long long)b * H + h) * S + s) * HD;
  float invf = __powf(10000.f, -(2.f * r) / (float)HD);
  float ang = (float)s * invf, cs = __cosf(ang), sn = __sinf(ang);
  float q1 = (float)ql[inoff + r], q2 = (float)ql[inoff + half + r];
  Q[outoff + r] = (__bf16)(q1 * cs - q2 * sn);
  Q[outoff + half + r] = (__bf16)(q2 * cs + q1 * sn);
  float k1 = (float)kl[inoff + r], k2 = (float)kl[inoff + half + r];
  K[outoff + r] = (__bf16)(k1 * cs - k2 * sn);
  K[outoff + half + r] = (__bf16)(k2 * cs + k1 * sn);
  V[outoff + r] = vl[inoff + r];
  V[outoff + half + r] = vl[inoff + half + r];
}

__global__ __launch_bounds__(256)
void softmax_kernel(const float* __restrict__ sc, __bf16* __restrict__ pr,
                    const unsigned char* __restrict__ mask, int S, int HS) {
  long long row = blockIdx.x;
  const float* p = sc + row * (long long)S;
  __bf16* q = pr + row * (long long)S;
  const unsigned char* mrow = mask + ((long long)(row / HS)) * S;
  const int tid = threadIdx.x;
  __shared__ float sh[8];
  float vals[4], mx = -3.4e38f;
#pragma unroll
  for (int j = 0; j < 4; ++j) {
    int c = tid + j * 256;
    float v = p[c] + (mrow[c] ? 0.f : -1e9f);
    vals[j] = v; mx = fmaxf(mx, v);
  }
  for (int o = 16; o; o >>= 1) mx = fmaxf(mx, __shfl_xor(mx, o, 32));
  if ((tid & 31) == 0) sh[tid >> 5] = mx;
  __syncthreads();
  if (tid == 0) { float m = sh[0]; for (int w = 1; w < 8; ++w) m = fmaxf(m, sh[w]); sh[0] = m; }
  __syncthreads();
  mx = sh[0];
  __syncthreads();
  float s = 0.f;
#pragma unroll
  for (int j = 0; j < 4; ++j) { vals[j] = __expf(vals[j] - mx); s += vals[j]; }
  for (int o = 16; o; o >>= 1) s += __shfl_xor(s, o, 32);
  if ((tid & 31) == 0) sh[tid >> 5] = s;
  __syncthreads();
  if (tid == 0) { float m = 0.f; for (int w = 0; w < 8; ++w) m += sh[w]; sh[0] = m; }
  __syncthreads();
  float inv = 1.f / sh[0];
#pragma unroll
  for (int j = 0; j < 4; ++j) q[tid + j * 256] = (__bf16)(vals[j] * inv);
}

__global__ void merge_heads_kernel(const float* __restrict__ O, __bf16* __restrict__ ctx,
                                   int Bsz, int S, int H, int HD) {
  long long i = (long long)blockIdx.x * 256 + threadIdx.x;
  long long total = (long long)Bsz * S * H * HD;
  if (i >= total) return;
  int d = (int)(i % HD); long long t = i / HD;
  int h = (int)(t % H); t /= H;
  int s = (int)(t % S); int b = (int)(t / S);
  ctx[((long long)(b * S + s)) * (H * HD) + h * HD + d] =
      (__bf16)O[(((long long)b * H + h) * S + s) * HD + d];
}

__device__ float block_reduce_sum(float v) {
  __shared__ float sh[9];
  const int tid = threadIdx.x;
  for (int o = 16; o; o >>= 1) v += __shfl_xor(v, o, 32);
  __syncthreads();
  if ((tid & 31) == 0) sh[tid >> 5] = v;
  __syncthreads();
  if (tid == 0) { float s = 0.f; for (int w = 0; w < 8; ++w) s += sh[w]; sh[8] = s; }
  __syncthreads();
  float r = sh[8];
  __syncthreads();
  return r;
}

__global__ __launch_bounds__(256)
void add_ln_kernel(float* __restrict__ x, __bf16* __restrict__ xb,
                   const float* __restrict__ o, const float* __restrict__ g,
                   const float* __restrict__ b, int D) {
  long long t = blockIdx.x;
  float* xr = x + t * D;
  __bf16* xbr = xb + t * D;
  const float* orow = o + t * D;
  const int tid = threadIdx.x;
  float v[4], s = 0.f;
#pragma unroll
  for (int j = 0; j < 4; ++j) { int c = tid + j * 256; v[j] = xr[c] + orow[c]; s += v[j]; }
  float mu = block_reduce_sum(s) / (float)D;
  float q = 0.f;
#pragma unroll
  for (int j = 0; j < 4; ++j) { float d = v[j] - mu; q += d * d; }
  float rstd = rsqrtf(block_reduce_sum(q) / (float)D + 1e-5f);
#pragma unroll
  for (int j = 0; j < 4; ++j) {
    int c = tid + j * 256;
    float r = (v[j] - mu) * rstd * g[c] + b[c];
    xr[c] = r;
    xbr[c] = (__bf16)r;
  }
}

__global__ __launch_bounds__(256)
void cls_kernel(const float* __restrict__ X, const float* __restrict__ g,
                const float* __restrict__ bln, const float* __restrict__ Wc,
                const float* __restrict__ bc, float* __restrict__ out,
                int SD, int D, int Bsz, int C) {
  const int tid = threadIdx.x;
  for (int b = 0; b < Bsz; ++b) {
    const float* xr = X + (long long)b * SD;   // token 0 of batch b
    float v[4], s = 0.f;
#pragma unroll
    for (int j = 0; j < 4; ++j) { int c = tid + j * 256; v[j] = xr[c]; s += v[j]; }
    float mu = block_reduce_sum(s) / (float)D;
    float q = 0.f;
#pragma unroll
    for (int j = 0; j < 4; ++j) { float d = v[j] - mu; q += d * d; }
    float rstd = rsqrtf(block_reduce_sum(q) / (float)D + 1e-5f);
    for (int cc = 0; cc < C; ++cc) {
      float dacc = 0.f;
#pragma unroll
      for (int j = 0; j < 4; ++j) {
        int c = tid + j * 256;
        float nv = (v[j] - mu) * rstd * g[c] + bln[c];
        dacc += nv * Wc[(long long)cc * D + c];
      }
      float tot = block_reduce_sum(dacc);
      if (tid == 0) out[b * C + cc] = tot + bc[cc];
    }
  }
}

// ---------------- host orchestration ----------------
extern "C" void kernel_launch(void* const* d_in, const int* in_sizes, int n_in,
                              void* d_out, int out_size, void* d_ws, size_t ws_size,
                              hipStream_t stream) {
  (void)in_sizes; (void)n_in; (void)out_size; (void)ws_size;
  constexpr int Vv = 32000, D = 1024, Lyr = 12, H = 16, DFF = 4096, S = 1024,
                Bsz = 2, C = 2, HD = D / H;
  constexpr int NT = Bsz * S;  // 2048 tokens

  const int* x_t = (const int*)d_in[0];
  const unsigned char* mask = (const unsigned char*)d_in[1];
  const float* emb = (const float*)d_in[2];
  const float* Wq = (const float*)d_in[3];
  const float* Wk = (const float*)d_in[4];
  const float* Wv = (const float*)d_in[5];
  const float* Wo = (const float*)d_in[6];
  const float* bq = (const float*)d_in[7];
  const float* bk = (const float*)d_in[8];
  const float* bv = (const float*)d_in[9];
  const float* bo = (const float*)d_in[10];
  const float* ln1g = (const float*)d_in[11];
  const float* ln1b = (const float*)d_in[12];
  const float* ln2g = (const float*)d_in[13];
  const float* ln2b = (const float*)d_in[14];
  const float* W1 = (const float*)d_in[15];
  const float* b1 = (const float*)d_in[16];
  const float* W2 = (const float*)d_in[17];
  const float* b2 = (const float*)d_in[18];
  const float* Wvoc = (const float*)d_in[19];
  const float* bvoc = (const float*)d_in[20];
  const float* lncg = (const float*)d_in[21];
  const float* lncb = (const float*)d_in[22];
  const float* Wcls = (const float*)d_in[23];
  const float* bcls = (const float*)d_in[24];
  float* out = (float*)d_out;

  // workspace carve-up (256B aligned)
  char* cur = (char*)d_ws;
  auto alloc = [&](size_t bytes) -> void* {
    void* p = cur;
    cur += (bytes + 255) & ~(size_t)255;
    return p;
  };
  float*  X     = (float*)alloc((size_t)NT * D * 4);
  float*  proj  = (float*)alloc((size_t)NT * D * 4);
  float*  Oh    = (float*)alloc((size_t)NT * D * 4);
  float*  scores = (float*)alloc((size_t)Bsz * H * S * S * 4);
  __bf16* Xb    = (__bf16*)alloc((size_t)NT * D * 2);
  __bf16* qlin  = (__bf16*)alloc((size_t)NT * D * 2);
  __bf16* klin  = (__bf16*)alloc((size_t)NT * D * 2);
  __bf16* vlin  = (__bf16*)alloc((size_t)NT * D * 2);
  __bf16* Qh    = (__bf16*)alloc((size_t)NT * D * 2);
  __bf16* Kh    = (__bf16*)alloc((size_t)NT * D * 2);
  __bf16* Vh    = (__bf16*)alloc((size_t)NT * D * 2);
  __bf16* ctxb  = (__bf16*)alloc((size_t)NT * D * 2);
  __bf16* ffhb  = (__bf16*)alloc((size_t)NT * DFF * 2);
  __bf16* probs = (__bf16*)alloc((size_t)Bsz * H * S * S * 2);
  __bf16* Wqb   = (__bf16*)alloc((size_t)Lyr * D * D * 2);
  __bf16* Wkb   = (__bf16*)alloc((size_t)Lyr * D * D * 2);
  __bf16* Wvb   = (__bf16*)alloc((size_t)Lyr * D * D * 2);
  __bf16* Wob   = (__bf16*)alloc((size_t)Lyr * D * D * 2);
  __bf16* W1b   = (__bf16*)alloc((size_t)Lyr * DFF * D * 2);
  __bf16* W2b   = (__bf16*)alloc((size_t)Lyr * D * DFF * 2);
  __bf16* Wvocb = (__bf16*)alloc((size_t)Vv * D * 2);

  auto cvt = [&](const float* src, __bf16* dst, long long n) {
    cvt_bf16_kernel<<<dim3((unsigned)((n + 1023) / 1024)), dim3(256), 0, stream>>>(src, dst, n);
  };
  // one-time (per launch) bf16 weight conversion
  cvt(Wq, Wqb, (long long)Lyr * D * D);
  cvt(Wk, Wkb, (long long)Lyr * D * D);
  cvt(Wv, Wvb, (long long)Lyr * D * D);
  cvt(Wo, Wob, (long long)Lyr * D * D);
  cvt(W1, W1b, (long long)Lyr * DFF * D);
  cvt(W2, W2b, (long long)Lyr * D * DFF);
  cvt(Wvoc, Wvocb, (long long)Vv * D);

  auto gemm = [&](const __bf16* A, const __bf16* Bm, const float* bias, float* OutF,
                  __bf16* OutH, int M, int N, int K, int batch, long long sA,
                  long long sB, long long sO, float scale, bool relu, bool transb) {
    dim3 grid((N + 127) / 128, (M + 127) / 128, batch), blk(256);
    if (transb)
      gemm_wmma<true, false><<<grid, blk, 0, stream>>>(A, Bm, bias, OutF, OutH, M, N, K, sA, sB, sO, scale);
    else if (relu)
      gemm_wmma<false, true><<<grid, blk, 0, stream>>>(A, Bm, bias, OutF, OutH, M, N, K, sA, sB, sO, scale);
    else
      gemm_wmma<false, false><<<grid, blk, 0, stream>>>(A, Bm, bias, OutF, OutH, M, N, K, sA, sB, sO, scale);
  };

  // embedding gather (fp32 residual + bf16 activation)
  {
    long long total = (long long)NT * D;
    embed_kernel<<<dim3((unsigned)((total + 255) / 256)), dim3(256), 0, stream>>>(
        x_t, emb, X, Xb, total, D);
  }

  const float iscale = 1.f / sqrtf((float)HD);
  for (int i = 0; i < Lyr; ++i) {
    // QKV projections: x @ W.T + b  (bf16 out, consumed by RoPE)
    gemm(Xb, Wqb + (size_t)i * D * D, bq + (size_t)i * D, nullptr, qlin, NT, D, D,
         1, 0, 0, 0, 1.f, false, false);
    gemm(Xb, Wkb + (size_t)i * D * D, bk + (size_t)i * D, nullptr, klin, NT, D, D,
         1, 0, 0, 0, 1.f, false, false);
    gemm(Xb, Wvb + (size_t)i * D * D, bv + (size_t)i * D, nullptr, vlin, NT, D, D,
         1, 0, 0, 0, 1.f, false, false);
    // RoPE + split to head-major [B,H,S,HD] (bf16)
    {
      long long total = (long long)Bsz * S * H * (HD / 2);
      rope_split_kernel<<<dim3((unsigned)((total + 255) / 256)), dim3(256), 0, stream>>>(
          qlin, klin, vlin, Qh, Kh, Vh, Bsz, S, H, HD);
    }
    // scores = Q @ K^T * (1/sqrt(HD)), batched over B*H (fp32 out for softmax)
    gemm(Qh, Kh, nullptr, scores, nullptr, S, S, HD, Bsz * H,
         (long long)S * HD, (long long)S * HD, (long long)S * S, iscale, false, false);
    softmax_kernel<<<dim3(Bsz * H * S), dim3(256), 0, stream>>>(scores, probs, mask, S, H * S);
    // O = attn @ V (B stored [K,N]); fp32 out
    gemm(probs, Vh, nullptr, Oh, nullptr, S, HD, S, Bsz * H,
         (long long)S * S, (long long)S * HD, (long long)S * HD, 1.f, false, true);
    {
      long long total = (long long)NT * D;
      merge_heads_kernel<<<dim3((unsigned)((total + 255) / 256)), dim3(256), 0, stream>>>(
          Oh, ctxb, Bsz, S, H, HD);
    }
    gemm(ctxb, Wob + (size_t)i * D * D, bo + (size_t)i * D, proj, nullptr, NT, D, D,
         1, 0, 0, 0, 1.f, false, false);
    add_ln_kernel<<<dim3(NT), dim3(256), 0, stream>>>(X, Xb, proj, ln1g + (size_t)i * D,
                                                      ln1b + (size_t)i * D, D);
    // FFN: relu(x@W1.T+b1) in bf16, then @W2.T+b2 in fp32
    gemm(Xb, W1b + (size_t)i * DFF * D, b1 + (size_t)i * DFF, nullptr, ffhb, NT, DFF, D,
         1, 0, 0, 0, 1.f, true, false);
    gemm(ffhb, W2b + (size_t)i * D * DFF, b2 + (size_t)i * D, proj, nullptr, NT, D, DFF,
         1, 0, 0, 0, 1.f, false, false);
    add_ln_kernel<<<dim3(NT), dim3(256), 0, stream>>>(X, Xb, proj, ln2g + (size_t)i * D,
                                                      ln2b + (size_t)i * D, D);
  }

  // vocab logits straight into d_out
  gemm(Xb, Wvocb, bvoc, out, nullptr, NT, Vv, D, 1, 0, 0, 0, 1.f, false, false);
  // classifier head on token 0 of each batch
  cls_kernel<<<dim3(1), dim3(256), 0, stream>>>(X, lncg, lncb, Wcls, bcls,
                                                out + (size_t)NT * Vv, S * D, D, Bsz, C);
}
